// WASAM_26568667693531
// MI455X (gfx1250) — matrix operand order repaired and checked by reading, not compiled, not run on hardware
//
#include <hip/hip_runtime.h>
#include <hip/hip_bf16.h>

// MI455X / gfx1250 dual-branch row-attention block.
//
// Roofline: ~35 GFLOP vs ~192 MB unavoidable HBM traffic (AI ~182 FLOP/B).
// Memory floor at 23.3 TB/s ~ 8us; with f16 WMMA (16x16x32, f32 accum) compute
// sits at/below that floor (f32 WMMA 16x16x4 would be ~8x slower -> compute
// bound). So: convert activations to f16 once in LDS, run every GEMM on
// v_wmma_f32_16x16x32_f16, keep the whole per-(b,h) working set (X, Q, V,
// 256x256 scores) resident in the 320KB WGP LDS.
//
// This revision cleans the instruction mix:
//  * left/right work split into separate uniform loops (no exec-mask branching
//    in GEMM epilogues),
//  * ALL WMMA operands are K-contiguous and 16B-aligned in LDS (strides 72 /
//    264 halves, multiples of 8 halves, row offsets ~ 4 words mod 64 banks), so
//    every fragment load is two ds_load_b128 instead of 16x ds_load_u16 +
//    v_mov_b16 repacking. Weights stay in natural [o][c] order; V is stored
//    transposed [c][v]; Q_r is consumed transposed.
//
// One workgroup (256 threads = 8 wave32 waves) per (b,h) row; grid = 4*256.

typedef __attribute__((ext_vector_type(16))) _Float16 v16h;
typedef __attribute__((ext_vector_type(8)))  _Float16 v8h;
typedef __attribute__((ext_vector_type(8)))  float    v8f;

#define DEVINL __device__ __forceinline__

constexpr int C     = 64;    // C_RGB == CR
constexpr int C2    = 32;
constexpr int W     = 256;
constexpr int H     = 256;
constexpr int BATCH = 4;
constexpr int HW    = H * W; // 65536

// ---- LDS strides (halves). All multiples of 8 halves (16B) for b128 loads;
// row byte-offsets are 4 words mod 64 banks -> conflict-free fragment loads.
constexpr int SX   = 72;     // [256 x 64] activation / Q arrays
constexpr int SSTR = 264;    // 256x256 score matrix
constexpr int SVT  = 264;    // transposed V arrays [32 c][256 v]

// ---- LDS layout (bytes); every array base 16B-aligned ----
constexpr int OFF_XL   = 0;        // raw left row f16  [256*72]h = 36864B
constexpr int OFF_XR   = 36864;    //                              36864B
constexpr int OFF_WL1G = 73728;    // (ln_w .* w_l1) [o][c] f16  64*72h = 9216B
constexpr int OFF_WR1T = 82944;    // w_r1 [o][c]                       9216B
constexpr int OFF_WL2T = 92160;    // w_l2 [o][c]  32*72h = 4608B
constexpr int OFF_WR2T = 96768;
constexpr int OFF_WLRT = 101376;
constexpr int OFF_WRRT = 105984;
constexpr int OFF_MU   = 110592;   // f32[256]
constexpr int OFF_RSTD = 111616;   // f32[256]
constexpr int OFF_T1   = 112640;   // f32[64]  sum_c g_c*W1[o,c]
constexpr int OFF_T2   = 112896;   // f32[64]  sum_c lnb_c*W1[o,c] + b_l1[o]
constexpr int OFF_BR1  = 113152;   // f32[64]
constexpr int OFF_BL2  = 113408;   // f32[32]
constexpr int OFF_BR2  = 113536;
constexpr int OFF_BLR  = 113664;
constexpr int OFF_BRR  = 113792;
constexpr int OFF_BETA = 113920;
constexpr int OFF_GAMMA= 114048;
constexpr int OFF_RMAX = 114176;   // f32[256] row max of S
constexpr int OFF_RINV = 115200;   // f32[256] 1/row-sumexp
constexpr int OFF_CMAX = 116224;   // f32[256] col max
constexpr int OFF_CINV = 117248;   // f32[256] 1/col-sumexp -> end 118272
constexpr int OFF_QL   = 118272;   // Q_l f16 [256*72]h = 36864B (dead after S)
constexpr int OFF_QR   = 155136;   //                             (dead after S)
constexpr int OFF_S    = 192000;   // scores f16 [256*264]h = 135168B -> 327168
// overlays on the dead Q region [118272, 192000):
constexpr int OFF_VLT  = 118272;   // V_l^T f16 [32*264]h = 16896B
constexpr int OFF_VRT  = 135168;   //                       16896B -> 152064
constexpr int OFF_OSTG = 152064;   // f32[32][256] staging = 32768B -> 184832

constexpr int LDS_BYTES = 327168;  // <= 320KB (327680) CDNA5 per-WG LDS cap

// ------------- WMMA fragment helpers (ISA 7.12.2 layouts, wave32) -------------

DEVINL v16h cat16(v8h lo, v8h hi) {
  return __builtin_shufflevector(lo, hi, 0, 1, 2, 3, 4, 5, 6, 7,
                                         8, 9, 10, 11, 12, 13, 14, 15);
}

// A 16x32 f16 tile from row-major [m][k] array.
// lane l<16 -> M=l, halves 0..7: K=hi*8+0..7, halves 8..15: K=16+hi*8+0..7
// -> two aligned 16B contiguous chunks per lane.
DEVINL v16h load_A(const _Float16* base, int stride, int m0, int k0, int lane) {
  const int l = lane & 15, hi = lane >> 4;
  const _Float16* row = base + (m0 + l) * stride + k0 + hi * 8;
  return cat16(*(const v8h*)row, *(const v8h*)(row + 16));
}

// B 32x16 f16 tile where logical B[k][n] is stored K-contiguous as src[n][k]:
// lane -> N=lane&15, halves 0..7: K=hi*8+0..7, halves 8..15: K=16+hi*8+0..7.
DEVINL v16h load_Bt(const _Float16* base, int stride, int k0, int n0, int lane) {
  const int n = lane & 15, hi = lane >> 4;
  const _Float16* p = base + (n0 + n) * stride + k0 + hi * 8;
  return cat16(*(const v8h*)p, *(const v8h*)(p + 16));
}

// A tile of P1 = row-softmax(S): element (m,k) = exp(S[m][k]-rmax[m])*rinv[m]
DEVINL v16h load_A_P1(const _Float16* S, const float* rmax, const float* rinv,
                      int m0, int k0, int lane) {
  const int l = lane & 15, hi = lane >> 4;
  const int m = m0 + l;
  const float mx = rmax[m], iv = rinv[m];
  const _Float16* p = S + m * SSTR + k0 + hi * 8;
  const v8h x0 = *(const v8h*)p, x1 = *(const v8h*)(p + 16);
  v16h a;
#pragma unroll
  for (int j = 0; j < 8; ++j) {
    a[j]     = (_Float16)(__expf((float)x0[j] - mx) * iv);
    a[8 + j] = (_Float16)(__expf((float)x1[j] - mx) * iv);
  }
  return a;
}

// A tile of P2 = col-softmax(S), S read transposed: (m=v,k=w) =
// exp(S[k][m]-cmax[m])*cinv[m]  (strided 2B gathers; unavoidable without a
// second 128KB transposed copy of S)
DEVINL v16h load_A_P2(const _Float16* S, const float* cmax, const float* cinv,
                      int m0, int k0, int lane) {
  const int l = lane & 15, hi = lane >> 4;
  const int m = m0 + l;
  const float mx = cmax[m], iv = cinv[m];
  const _Float16* p = S + (k0 + hi * 8) * SSTR + m;
  v16h a;
#pragma unroll
  for (int j = 0; j < 8; ++j) {
    a[j]     = (_Float16)(__expf((float)p[j * SSTR]        - mx) * iv);
    a[8 + j] = (_Float16)(__expf((float)p[(16 + j) * SSTR] - mx) * iv);
  }
  return a;
}

DEVINL v8f wmma_f16(v16h a, v16h b, v8f c) {
  return __builtin_amdgcn_wmma_f32_16x16x32_f16(false, a, false, b, (short)0, c,
                                                false, false);
}

// ---------------------------------- kernel ----------------------------------

__global__ __launch_bounds__(256, 1)
void wasam_row_attn(const float* __restrict__ x_rgb, const float* __restrict__ x_hfw,
                    const float* __restrict__ ln_w,  const float* __restrict__ ln_b,
                    const float* __restrict__ w_l1,  const float* __restrict__ b_l1,
                    const float* __restrict__ w_r1,  const float* __restrict__ b_r1,
                    const float* __restrict__ w_l2,  const float* __restrict__ b_l2,
                    const float* __restrict__ w_r2,  const float* __restrict__ b_r2,
                    const float* __restrict__ w_lres,const float* __restrict__ b_lres,
                    const float* __restrict__ w_rres,const float* __restrict__ b_rres,
                    const float* __restrict__ beta,  const float* __restrict__ gamma,
                    float* __restrict__ out) {
  extern __shared__ char smem[];
  _Float16* sXl   = (_Float16*)(smem + OFF_XL);
  _Float16* sXr   = (_Float16*)(smem + OFF_XR);
  _Float16* sWl1g = (_Float16*)(smem + OFF_WL1G);
  _Float16* sWr1t = (_Float16*)(smem + OFF_WR1T);
  _Float16* sWl2t = (_Float16*)(smem + OFF_WL2T);
  _Float16* sWr2t = (_Float16*)(smem + OFF_WR2T);
  _Float16* sWlrt = (_Float16*)(smem + OFF_WLRT);
  _Float16* sWrrt = (_Float16*)(smem + OFF_WRRT);
  float* sMu   = (float*)(smem + OFF_MU);
  float* sRstd = (float*)(smem + OFF_RSTD);
  float* sT1   = (float*)(smem + OFF_T1);
  float* sT2   = (float*)(smem + OFF_T2);
  float* sBr1  = (float*)(smem + OFF_BR1);
  float* sBl2  = (float*)(smem + OFF_BL2);
  float* sBr2  = (float*)(smem + OFF_BR2);
  float* sBlr  = (float*)(smem + OFF_BLR);
  float* sBrr  = (float*)(smem + OFF_BRR);
  float* sBeta = (float*)(smem + OFF_BETA);
  float* sGam  = (float*)(smem + OFF_GAMMA);
  float* sRMax = (float*)(smem + OFF_RMAX);
  float* sRInv = (float*)(smem + OFF_RINV);
  float* sCMax = (float*)(smem + OFF_CMAX);
  float* sCInv = (float*)(smem + OFF_CINV);
  _Float16* sQl  = (_Float16*)(smem + OFF_QL);
  _Float16* sQr  = (_Float16*)(smem + OFF_QR);
  _Float16* sS   = (_Float16*)(smem + OFF_S);
  _Float16* sVlT = (_Float16*)(smem + OFF_VLT);  // overlays dead Q region
  _Float16* sVrT = (_Float16*)(smem + OFF_VRT);
  float*    sOut = (float*)(smem + OFF_OSTG);

  const int tid  = threadIdx.x;
  const int lane = tid & 31;
  const int wv   = tid >> 5;              // wave id 0..7
  const int bb   = blockIdx.x >> 8;       // batch
  const int hh   = blockIdx.x & (H - 1);  // row

  const v8f Z8 = {0.f, 0.f, 0.f, 0.f, 0.f, 0.f, 0.f, 0.f};

  // ---- Phase 0: stage weights (natural [o][c] order = K-contiguous for B) ----
  for (int i = tid; i < C * C; i += 256) {      // i = o*64 + c
    const int o = i >> 6, c = i & 63;
    sWl1g[o * SX + c] = (_Float16)(ln_w[c] * w_l1[i]);   // LN gain folded in
    sWr1t[o * SX + c] = (_Float16)w_r1[i];
  }
  for (int i = tid; i < C2 * C; i += 256) {     // i = o*64 + c, o < 32
    const int o = i >> 6, c = i & 63;
    sWl2t[o * SX + c] = (_Float16)w_l2[i];
    sWr2t[o * SX + c] = (_Float16)w_r2[i];
    sWlrt[o * SX + c] = (_Float16)w_lres[i];
    sWrrt[o * SX + c] = (_Float16)w_rres[i];
  }
  if (tid < C) {
    float t1 = 0.f, t2 = 0.f;
    for (int c = 0; c < C; ++c) {
      const float w1 = w_l1[tid * C + c];
      t1 += ln_w[c] * w1;
      t2 += ln_b[c] * w1;
    }
    sT1[tid] = t1;
    sT2[tid] = t2 + b_l1[tid];
    sBr1[tid] = b_r1[tid];
  }
  if (tid < C2) {
    sBl2[tid] = b_l2[tid]; sBr2[tid] = b_r2[tid];
    sBlr[tid] = b_lres[tid]; sBrr[tid] = b_rres[tid];
    sBeta[tid] = beta[tid]; sGam[tid] = gamma[tid];
  }

  // ---- Phase 1: load row, f32->f16, LayerNorm stats (coalesced per channel) ----
  {
    const float* xl = x_rgb + (size_t)bb * C * HW + (size_t)hh * W + tid;
    float s = 0.f, s2 = 0.f;
    for (int c = 0; c < C; ++c) {
      const float f = xl[(size_t)c * HW];
      sXl[tid * SX + c] = (_Float16)f;
      s += f; s2 += f * f;
    }
    const float mu = s * (1.f / C);
    sMu[tid] = mu;
    sRstd[tid] = rsqrtf(fmaxf(s2 * (1.f / C) - mu * mu, 0.f) + 1e-6f);
    const float* xr = x_hfw + (size_t)bb * C * HW + (size_t)hh * W + tid;
    for (int c = 0; c < C; ++c) sXr[tid * SX + c] = (_Float16)xr[(size_t)c * HW];
  }
  __syncthreads();

  // ---- Phase 2a: Q_l = LN(X_l) W_l1^T + b_l1 (LN folded: affine fixup) ----
  for (int t = wv; t < 64; t += 8) {
    const int m0 = (t >> 2) * 16, n0 = (t & 3) * 16;
    v8f acc = Z8;
    acc = wmma_f16(load_A(sXl, SX, m0, 0,  lane), load_Bt(sWl1g, SX, 0,  n0, lane), acc);
    acc = wmma_f16(load_A(sXl, SX, m0, 32, lane), load_Bt(sWl1g, SX, 32, n0, lane), acc);
    const int n = n0 + (lane & 15), hi = lane >> 4;
    const float t1 = sT1[n], t2 = sT2[n];
#pragma unroll
    for (int r = 0; r < 8; ++r) {
      const int m = m0 + r + 8 * hi;
      const float rs = sRstd[m];
      sQl[m * SX + n] = (_Float16)(rs * acc[r] - rs * sMu[m] * t1 + t2);
    }
  }
  // ---- Phase 2b: Q_r = X_r W_r1^T + b_r1 ----
  for (int t = wv; t < 64; t += 8) {
    const int m0 = (t >> 2) * 16, n0 = (t & 3) * 16;
    v8f acc = Z8;
    acc = wmma_f16(load_A(sXr, SX, m0, 0,  lane), load_Bt(sWr1t, SX, 0,  n0, lane), acc);
    acc = wmma_f16(load_A(sXr, SX, m0, 32, lane), load_Bt(sWr1t, SX, 32, n0, lane), acc);
    const int n = n0 + (lane & 15), hi = lane >> 4;
    const float bv = sBr1[n];
#pragma unroll
    for (int r = 0; r < 8; ++r)
      sQr[(m0 + r + 8 * hi) * SX + n] = (_Float16)(acc[r] + bv);
  }
  __syncthreads();

  // ---- Phase 3: S = (Q_l Q_r^T) / 8   (256 tiles; Q_r consumed transposed) ----
  for (int t = wv; t < 256; t += 8) {
    const int m0 = (t >> 4) * 16, n0 = (t & 15) * 16;
    v8f acc = Z8;
    acc = wmma_f16(load_A(sQl, SX, m0, 0,  lane), load_Bt(sQr, SX, 0,  n0, lane), acc);
    acc = wmma_f16(load_A(sQl, SX, m0, 32, lane), load_Bt(sQr, SX, 32, n0, lane), acc);
    const int n = n0 + (lane & 15), hi = lane >> 4;
#pragma unroll
    for (int r = 0; r < 8; ++r)
      sS[(m0 + r + 8 * hi) * SSTR + n] = (_Float16)(acc[r] * 0.125f);
  }
  __syncthreads();

  // ---- Phase 3b: softmax stats over rows AND columns of S; V projections ----
  {
    const v8h* row = (const v8h*)(sS + tid * SSTR);   // 16B-aligned vector scans
    float mx = -3.0e38f;
    for (int q = 0; q < W / 8; ++q) {
      const v8h x = row[q];
#pragma unroll
      for (int j = 0; j < 8; ++j) mx = fmaxf(mx, (float)x[j]);
    }
    float sum = 0.f;
    for (int q = 0; q < W / 8; ++q) {
      const v8h x = row[q];
#pragma unroll
      for (int j = 0; j < 8; ++j) sum += __expf((float)x[j] - mx);
    }
    sRMax[tid] = mx; sRInv[tid] = 1.f / sum;

    const _Float16* col = sS + tid;                   // strided column scan
    mx = -3.0e38f;
    for (int w2 = 0; w2 < W; ++w2) mx = fmaxf(mx, (float)col[w2 * SSTR]);
    sum = 0.f;
    for (int w2 = 0; w2 < W; ++w2) sum += __expf((float)col[w2 * SSTR] - mx);
    sCMax[tid] = mx; sCInv[tid] = 1.f / sum;
  }
  // V_l^T (stored [c][v] so later B loads are contiguous)
  for (int t = wv; t < 32; t += 8) {
    const int m0 = (t >> 1) * 16, n0 = (t & 1) * 16;
    v8f acc = Z8;
    acc = wmma_f16(load_A(sXl, SX, m0, 0,  lane), load_Bt(sWl2t, SX, 0,  n0, lane), acc);
    acc = wmma_f16(load_A(sXl, SX, m0, 32, lane), load_Bt(sWl2t, SX, 32, n0, lane), acc);
    const int n = n0 + (lane & 15), hi = lane >> 4;
    const float bv = sBl2[n];
#pragma unroll
    for (int r = 0; r < 8; ++r)
      sVlT[n * SVT + (m0 + r + 8 * hi)] = (_Float16)(acc[r] + bv);
  }
  // V_r^T
  for (int t = wv; t < 32; t += 8) {
    const int m0 = (t >> 1) * 16, n0 = (t & 1) * 16;
    v8f acc = Z8;
    acc = wmma_f16(load_A(sXr, SX, m0, 0,  lane), load_Bt(sWr2t, SX, 0,  n0, lane), acc);
    acc = wmma_f16(load_A(sXr, SX, m0, 32, lane), load_Bt(sWr2t, SX, 32, n0, lane), acc);
    const int n = n0 + (lane & 15), hi = lane >> 4;
    const float bv = sBr2[n];
#pragma unroll
    for (int r = 0; r < 8; ++r)
      sVrT[n * SVT + (m0 + r + 8 * hi)] = (_Float16)(acc[r] + bv);
  }
  __syncthreads();

  // ---- Phase 4: out_l = X_l Wlres^T + b + beta * (rowsoftmax(S) V_r) ----
  for (int t = wv; t < 32; t += 8) {
    const int m0 = (t >> 1) * 16, n0 = (t & 1) * 16;
    v8f res = Z8;
    res = wmma_f16(load_A(sXl, SX, m0, 0,  lane), load_Bt(sWlrt, SX, 0,  n0, lane), res);
    res = wmma_f16(load_A(sXl, SX, m0, 32, lane), load_Bt(sWlrt, SX, 32, n0, lane), res);
    v8f f = Z8;
    for (int k0 = 0; k0 < W; k0 += 32)   // softmax applied on the fly in A frags
      f = wmma_f16(load_A_P1(sS, sRMax, sRInv, m0, k0, lane),
                   load_Bt(sVrT, SVT, k0, n0, lane), f);
    const int n = n0 + (lane & 15), hi = lane >> 4;
    const float bv = sBlr[n], sc = sBeta[n];
#pragma unroll
    for (int r = 0; r < 8; ++r)
      sOut[n * W + (m0 + r + 8 * hi)] = res[r] + bv + sc * f[r];
  }
  __syncthreads();
  for (int c = 0; c < C2; ++c)           // coalesced channel-row stores, ch 0..31
    out[((size_t)bb * C + c) * HW + (size_t)hh * W + tid] = sOut[c * W + tid];
  __syncthreads();

  // ---- Phase 5: out_r = X_r Wrres^T + b + gamma * (colsoftmax(S)^T V_l) ----
  for (int t = wv; t < 32; t += 8) {
    const int m0 = (t >> 1) * 16, n0 = (t & 1) * 16;
    v8f res = Z8;
    res = wmma_f16(load_A(sXr, SX, m0, 0,  lane), load_Bt(sWrrt, SX, 0,  n0, lane), res);
    res = wmma_f16(load_A(sXr, SX, m0, 32, lane), load_Bt(sWrrt, SX, 32, n0, lane), res);
    v8f f = Z8;
    for (int k0 = 0; k0 < W; k0 += 32)
      f = wmma_f16(load_A_P2(sS, sCMax, sCInv, m0, k0, lane),
                   load_Bt(sVlT, SVT, k0, n0, lane), f);
    const int n = n0 + (lane & 15), hi = lane >> 4;
    const float bv = sBrr[n], sc = sGam[n];
#pragma unroll
    for (int r = 0; r < 8; ++r)
      sOut[n * W + (m0 + r + 8 * hi)] = res[r] + bv + sc * f[r];
  }
  __syncthreads();
  for (int c = 0; c < C2; ++c)           // channels 32..63
    out[((size_t)bb * C + C2 + c) * HW + (size_t)hh * W + tid] = sOut[c * W + tid];
}

// ---------------------------------- launch ----------------------------------

extern "C" void kernel_launch(void* const* d_in, const int* in_sizes, int n_in,
                              void* d_out, int out_size, void* d_ws, size_t ws_size,
                              hipStream_t stream) {
  (void)in_sizes; (void)n_in; (void)d_ws; (void)ws_size; (void)out_size;

  // ~319.5KB dynamic LDS per workgroup (CDNA5 WGP cap = 320KB); raise the cap.
  (void)hipFuncSetAttribute((const void*)wasam_row_attn,
                            hipFuncAttributeMaxDynamicSharedMemorySize, LDS_BYTES);

  const float* x_rgb = (const float*)d_in[0];
  const float* x_hfw = (const float*)d_in[1];
  const float* ln_w  = (const float*)d_in[2];
  const float* ln_b  = (const float*)d_in[3];
  const float* w_l1  = (const float*)d_in[4];
  const float* b_l1  = (const float*)d_in[5];
  const float* w_r1  = (const float*)d_in[6];
  const float* b_r1  = (const float*)d_in[7];
  const float* w_l2  = (const float*)d_in[8];
  const float* b_l2  = (const float*)d_in[9];
  const float* w_r2  = (const float*)d_in[10];
  const float* b_r2  = (const float*)d_in[11];
  const float* w_lres= (const float*)d_in[12];
  const float* b_lres= (const float*)d_in[13];
  const float* w_rres= (const float*)d_in[14];
  const float* b_rres= (const float*)d_in[15];
  const float* beta  = (const float*)d_in[16];
  const float* gamma = (const float*)d_in[17];

  wasam_row_attn<<<dim3(BATCH * H), dim3(256), LDS_BYTES, stream>>>(
      x_rgb, x_hfw, ln_w, ln_b, w_l1, b_l1, w_r1, b_r1, w_l2, b_l2, w_r2, b_r2,
      w_lres, b_lres, w_rres, b_rres, beta, gamma, (float*)d_out);
}
